// NuvionPro_85607288143951
// MI455X (gfx1250) — compile-verified, matrix operand order checked by reading, not run on hardware
//
#include <hip/hip_runtime.h>
#include <hip/hip_bf16.h>

// MoE (top-2 of 8 experts), D=1024, F=2048, T=4096, fp32 in/out.
// Routed compute with bf16 WMMA (v_wmma_f32_16x16x32_bf16).
// fp32->bf16 conversion while staging into LDS; register double-buffering
// for converted tiles; gfx1250 async global->LDS DMA for bf16 activations.

#define E_   8
#define D_   1024
#define F_   2048
#define T_   4096

typedef __bf16 bf16;
typedef __attribute__((ext_vector_type(8)))  bf16  bf16x8;
typedef __attribute__((ext_vector_type(16))) bf16  v16bf;
typedef __attribute__((ext_vector_type(8)))  float v8f;
typedef __attribute__((ext_vector_type(4)))  float f32x4;

// ---- workspace layout (bytes) ----
// 0      : int cnt[8]
// 64     : int seg[9]
// 128    : int tok[E*T]
// 131200 : float gate[E*T]
// 524288 : bf16 act[8192][F]   (33.5 MB)
#define WS_TOKOFF   128
#define WS_GATEOFF  (128 + E_*T_*4)
#define WS_ACTOFF   (512*1024)

__device__ inline v16bf cat8(bf16x8 lo, bf16x8 hi) {
  return __builtin_shufflevector(lo, hi, 0,1,2,3,4,5,6,7,8,9,10,11,12,13,14,15);
}

__device__ inline v8f wmma_bf16(v16bf a, v16bf b, v8f c) {
  return __builtin_amdgcn_wmma_f32_16x16x32_bf16(false, a, false, b, (short)0, c,
                                                 false, false);
}

// 16x32 bf16 fragment from an LDS tile, row stride 40 elems (80B: 16B-aligned,
// bank-conflict-free). lanes 0-15: K 0-7/16-23, lanes 16-31: K 8-15/24-31.
__device__ inline v16bf load_frag(const bf16* base, int idx, int half) {
  const bf16* p = base + idx * 40;
  bf16x8 lo = *(const bf16x8*)(p + half * 8);
  bf16x8 hi = *(const bf16x8*)(p + 16 + half * 8);
  return cat8(lo, hi);
}

__device__ inline void cvt16(const f32x4 v0, const f32x4 v1, const f32x4 v2,
                             const f32x4 v3, bf16x8& o0, bf16x8& o1) {
#pragma unroll
  for (int i = 0; i < 4; ++i) {
    o0[i]     = (bf16)v0[i];
    o0[4 + i] = (bf16)v1[i];
    o1[i]     = (bf16)v2[i];
    o1[4 + i] = (bf16)v3[i];
  }
}

// gfx1250 async DMA: 16B per lane, global -> LDS, tracked by ASYNCcnt.
__device__ inline void async_ld_b128(unsigned lds_addr, const bf16* g) {
  asm volatile("global_load_async_to_lds_b128 %0, %1, off"
               :: "v"(lds_addr), "v"(g)
               : "memory");
}

// ---------------------------------------------------------------------------
// Kernel 0: zero output accumulator + routing counters
// ---------------------------------------------------------------------------
__global__ __launch_bounds__(256) void moe_init(float* __restrict__ out, int n4,
                                                int* __restrict__ cnt) {
  int i = blockIdx.x * 256 + threadIdx.x;
  if (blockIdx.x == 0 && threadIdx.x < 32) cnt[threadIdx.x] = 0;
  if (i < n4) {
    f32x4 z = {0.f, 0.f, 0.f, 0.f};
    ((f32x4*)out)[i] = z;
  }
}

// ---------------------------------------------------------------------------
// Kernel 1: router — one wave32 per token
// ---------------------------------------------------------------------------
__global__ __launch_bounds__(256) void moe_router(
    const float* __restrict__ x, const float* __restrict__ Wr,
    const float* __restrict__ temp, int* __restrict__ cnt,
    int* __restrict__ tok, float* __restrict__ gate) {
  const int wave = threadIdx.x >> 5;
  const int lane = threadIdx.x & 31;
  const int t    = blockIdx.x * 8 + wave;

  float acc[E_];
#pragma unroll
  for (int e = 0; e < E_; ++e) acc[e] = 0.f;

  const float* xr = x + (size_t)t * D_;
  for (int i = lane; i < D_; i += 32) {
    float xv = xr[i];
#pragma unroll
    for (int e = 0; e < E_; ++e) acc[e] += xv * Wr[e * D_ + i];
  }
#pragma unroll
  for (int e = 0; e < E_; ++e) {
#pragma unroll
    for (int off = 16; off > 0; off >>= 1)
      acc[e] += __shfl_xor(acc[e], off, 32);
  }

  if (lane == 0) {
    float tmp = temp[0];
    int e0 = 0;
    float l0 = acc[0];
#pragma unroll
    for (int e = 1; e < E_; ++e)
      if (acc[e] > l0) { l0 = acc[e]; e0 = e; }
    int e1 = -1;
    float l1 = -3.4e38f;
#pragma unroll
    for (int e = 0; e < E_; ++e)
      if (e != e0 && acc[e] > l1) { l1 = acc[e]; e1 = e; }
    float p0 = 1.f / (1.f + __expf((l1 - l0) / tmp));
    float p1 = 1.f - p0;
    int q0 = atomicAdd(&cnt[e0], 1);
    tok[e0 * T_ + q0]  = t;
    gate[e0 * T_ + q0] = p0;
    int q1 = atomicAdd(&cnt[e1], 1);
    tok[e1 * T_ + q1]  = t;
    gate[e1 * T_ + q1] = p1;
  }
}

// ---------------------------------------------------------------------------
// Kernel 2: segment offsets
// ---------------------------------------------------------------------------
__global__ void moe_seg(const int* __restrict__ cnt, int* __restrict__ seg) {
  if (threadIdx.x == 0) {
    int s = 0;
#pragma unroll
    for (int e = 0; e < E_; ++e) { seg[e] = s; s += cnt[e]; }
    seg[E_] = s;
  }
}

// ---------------------------------------------------------------------------
// Kernel 3: H = Xg @ W1[e]^T, SwiGLU -> packed bf16 activations
//   128 tokens x 64 act cols (128 B-cols: value + gate halves of W1).
//   Register double-buffering: next k-tile's global loads issued before the
//   WMMA chain of the current tile.
// ---------------------------------------------------------------------------
__global__ __launch_bounds__(256) void moe_ffn1(
    const float* __restrict__ x, const float* __restrict__ W1,
    const float* __restrict__ b1, const int* __restrict__ cnt,
    const int* __restrict__ seg, const int* __restrict__ tok,
    bf16* __restrict__ act) {
  const int e  = blockIdx.z;
  const int m0 = blockIdx.y * 128;
  const int n0 = blockIdx.x * 64;
  const int ne = cnt[e];
  if (m0 >= ne) return;

  __shared__ bf16 lA[128 * 40];
  __shared__ bf16 lB[128 * 40];
  __shared__ int  sTok[128];

  const int tid  = threadIdx.x;
  const int wv   = tid >> 5;
  const int lane = tid & 31;
  const int lrow = lane & 15;
  const int half = lane >> 4;
  const int ar   = tid >> 1;  // staging row/col 0..127
  const int ah   = tid & 1;   // staging k-half (16 floats)

  if (tid < 128) {
    int p = m0 + tid;
    sTok[tid] = (p < ne) ? tok[e * T_ + p] : -1;
  }
  __syncthreads();

  const float* W1e = W1 + (size_t)e * (2 * F_) * D_;
  const int tokraw = sTok[ar];
  const int toksafe = (tokraw < 0) ? 0 : tokraw;   // masked rows in epilogue
  const float* asrc0 = x + (size_t)toksafe * D_ + ah * 16;
  const int w1row = (ar < 64) ? (n0 + ar) : (F_ + n0 + (ar - 64));
  const float* bsrc0 = W1e + (size_t)w1row * D_ + ah * 16;

  v8f accU[4] = {};
  v8f accV[4] = {};

  f32x4 ra[4], rb[4];
#pragma unroll
  for (int i = 0; i < 4; ++i) {
    ra[i] = ((const f32x4*)asrc0)[i];
    rb[i] = ((const f32x4*)bsrc0)[i];
  }

  for (int k0 = 0; k0 < D_; k0 += 32) {
    __syncthreads();
    {
      bf16x8 o0, o1;
      cvt16(ra[0], ra[1], ra[2], ra[3], o0, o1);
      *(bf16x8*)(&lA[ar * 40 + ah * 16])     = o0;
      *(bf16x8*)(&lA[ar * 40 + ah * 16 + 8]) = o1;
      cvt16(rb[0], rb[1], rb[2], rb[3], o0, o1);
      *(bf16x8*)(&lB[ar * 40 + ah * 16])     = o0;
      *(bf16x8*)(&lB[ar * 40 + ah * 16 + 8]) = o1;
    }
    __syncthreads();

    if (k0 + 32 < D_) {  // prefetch next tile into registers (overlaps WMMAs)
      const float* an = asrc0 + k0 + 32;
      const float* bn = bsrc0 + k0 + 32;
      __builtin_prefetch(bn + 32, 0, 0);
#pragma unroll
      for (int i = 0; i < 4; ++i) {
        ra[i] = ((const f32x4*)an)[i];
        rb[i] = ((const f32x4*)bn)[i];
      }
    }

    // ---- compute: wave wv owns rows [wv*16, wv*16+16) ----
    v16bf afrag = load_frag(lA, wv * 16 + lrow, half);
    v16bf bu[4];
#pragma unroll
    for (int j = 0; j < 4; ++j) bu[j] = load_frag(lB, j * 16 + lrow, half);
#pragma unroll
    for (int j = 0; j < 4; ++j) accU[j] = wmma_bf16(afrag, bu[j], accU[j]);
    v16bf bv[4];
#pragma unroll
    for (int j = 0; j < 4; ++j) bv[j] = load_frag(lB, 64 + j * 16 + lrow, half);
#pragma unroll
    for (int j = 0; j < 4; ++j) accV[j] = wmma_bf16(afrag, bv[j], accV[j]);
  }

  // ---- SwiGLU epilogue: a = (u+bu) * silu(v+bv) ----
  const float* b1e  = b1 + (size_t)e * 2 * F_;
  const int    segb = seg[e];
#pragma unroll
  for (int j = 0; j < 4; ++j) {
    int   col = n0 + j * 16 + lrow;
    float bus = b1e[col];
    float bvs = b1e[F_ + col];
#pragma unroll
    for (int r = 0; r < 8; ++r) {
      int p = m0 + wv * 16 + half * 8 + r;
      if (p < ne) {
        float u = accU[j][r] + bus;
        float v = accV[j][r] + bvs;
        float a = u * (v / (1.f + __expf(-v)));
        act[(size_t)(segb + p) * F_ + col] = (bf16)a;
      }
    }
  }
}

// ---------------------------------------------------------------------------
// Kernel 4: Y = A @ W2[e]^T, out[tok] += (Y + b2) * gate
//   A tiles (bf16) DMA'd via global_load_async_to_lds_b128 into a
//   double-buffered LDS region (ASYNCcnt); W2 staged fp32->bf16 through regs.
// ---------------------------------------------------------------------------
__global__ __launch_bounds__(256) void moe_ffn2(
    const bf16* __restrict__ act, const float* __restrict__ W2,
    const float* __restrict__ b2, const int* __restrict__ cnt,
    const int* __restrict__ seg, const int* __restrict__ tok,
    const float* __restrict__ gate, float* __restrict__ out) {
  const int e  = blockIdx.z;
  const int m0 = blockIdx.y * 128;
  const int n0 = blockIdx.x * 64;
  const int ne = cnt[e];
  if (m0 >= ne) return;

  __shared__ bf16  lA[2][128 * 40];
  __shared__ bf16  lB[64 * 40];
  __shared__ int   sTok[128];
  __shared__ float sGate[128];

  const int tid  = threadIdx.x;
  const int wv   = tid >> 5;
  const int lane = tid & 31;
  const int lrow = lane & 15;
  const int half = lane >> 4;
  const int ar   = tid >> 1;  // A staging row 0..127
  const int ah   = tid & 1;
  const int bc   = tid >> 2;  // B staging col 0..63
  const int bq   = tid & 3;   // 8-float k chunk

  if (tid < 128) {
    int p = m0 + tid;
    sTok[tid]  = (p < ne) ? tok[e * T_ + p] : -1;
    sGate[tid] = (p < ne) ? gate[e * T_ + p] : 0.f;
  }
  __syncthreads();

  const float* W2e  = W2 + (size_t)e * D_ * F_;
  const int    segb = seg[e];
  const int    rowidx = ((m0 + ar) < ne) ? (segb + m0 + ar) : segb;  // safe
  const bf16*  agbase = act + (size_t)rowidx * F_ + ah * 16;
  const float* bsrc0  = W2e + (size_t)(n0 + bc) * F_ + bq * 8;

  v8f acc[4] = {};
  const int NK = F_ / 32;

  // prologue: DMA A tile 0, load B tile 0 into registers
  {
    unsigned la = (unsigned)(uintptr_t)(&lA[0][ar * 40 + ah * 16]);
    async_ld_b128(la, agbase);
    async_ld_b128(la + 16, agbase + 8);
  }
  f32x4 rb0 = ((const f32x4*)bsrc0)[0];
  f32x4 rb1 = ((const f32x4*)bsrc0)[1];

  for (int kt = 0; kt < NK; ++kt) {
    const int cur = kt & 1;
    const bool hasnext = (kt + 1) < NK;
    __syncthreads();  // all waves done reading lB / lA[1-cur] from prev iter

    // stage B (fp32 -> bf16)
    {
      bf16x8 o;
#pragma unroll
      for (int i = 0; i < 4; ++i) {
        o[i]     = (bf16)rb0[i];
        o[4 + i] = (bf16)rb1[i];
      }
      *(bf16x8*)(&lB[bc * 40 + bq * 8]) = o;
    }
    // DMA next A tile into the other buffer
    if (hasnext) {
      const bf16* g = agbase + (kt + 1) * 32;
      unsigned la = (unsigned)(uintptr_t)(&lA[1 - cur][ar * 40 + ah * 16]);
      async_ld_b128(la, g);
      async_ld_b128(la + 16, g + 8);
      asm volatile("s_wait_asynccnt 0x2" ::: "memory");  // tile kt complete
    } else {
      asm volatile("s_wait_asynccnt 0x0" ::: "memory");
    }
    __syncthreads();

    if (hasnext) {  // next B tile into registers (overlaps WMMAs)
      const float* bn = bsrc0 + (kt + 1) * 32;
      __builtin_prefetch(bn + 32, 0, 0);
      rb0 = ((const f32x4*)bn)[0];
      rb1 = ((const f32x4*)bn)[1];
    }

    v16bf afrag = load_frag(lA[cur], wv * 16 + lrow, half);
    v16bf bf[4];
#pragma unroll
    for (int j = 0; j < 4; ++j) bf[j] = load_frag(lB, j * 16 + lrow, half);
#pragma unroll
    for (int j = 0; j < 4; ++j) acc[j] = wmma_bf16(afrag, bf[j], acc[j]);
  }

  // ---- epilogue: gate-scale + scatter-accumulate ----
  const float* b2e = b2 + (size_t)e * D_;
#pragma unroll
  for (int j = 0; j < 4; ++j) {
    int   col = n0 + j * 16 + lrow;
    float bb  = b2e[col];
#pragma unroll
    for (int r = 0; r < 8; ++r) {
      int pl = wv * 16 + half * 8 + r;
      int p  = m0 + pl;
      if (p < ne) {
        int   tk = sTok[pl];
        float g  = sGate[pl];
        atomicAdd(&out[(size_t)tk * D_ + col], (acc[j][r] + bb) * g);
      }
    }
  }
}

// ---------------------------------------------------------------------------
extern "C" void kernel_launch(void* const* d_in, const int* in_sizes, int n_in,
                              void* d_out, int out_size, void* d_ws,
                              size_t ws_size, hipStream_t stream) {
  const float* x    = (const float*)d_in[0];
  const float* Wr   = (const float*)d_in[1];
  const float* temp = (const float*)d_in[2];
  const float* W1   = (const float*)d_in[3];
  const float* b1   = (const float*)d_in[4];
  const float* W2   = (const float*)d_in[5];
  const float* b2   = (const float*)d_in[6];
  float*       out  = (float*)d_out;

  char*  ws   = (char*)d_ws;
  int*   cnt  = (int*)(ws + 0);
  int*   seg  = (int*)(ws + 64);
  int*   tok  = (int*)(ws + WS_TOKOFF);
  float* gate = (float*)(ws + WS_GATEOFF);
  bf16*  act  = (bf16*)(ws + WS_ACTOFF);

  const int n4 = out_size / 4;
  moe_init<<<(n4 + 255) / 256, 256, 0, stream>>>(out, n4, cnt);
  moe_router<<<T_ / 8, 256, 0, stream>>>(x, Wr, temp, cnt, tok, gate);
  moe_seg<<<1, 32, 0, stream>>>(cnt, seg);
  moe_ffn1<<<dim3(F_ / 64, T_ / 128, E_), 256, 0, stream>>>(x, W1, b1, cnt, seg,
                                                            tok, act);
  moe_ffn2<<<dim3(D_ / 64, T_ / 128, E_), 256, 0, stream>>>(act, W2, b2, cnt,
                                                            seg, tok, gate, out);
}
